// PModel_70772471103949
// MI455X (gfx1250) — matrix operand order repaired and checked by reading, not compiled, not run on hardware
//
#include <hip/hip_runtime.h>

#define VOCAB 32000
#define EMB   512
#define HID   1024
#define BSZ   32
#define SEQ   256
#define ROWS  (BSZ * SEQ)   // 8192

typedef __bf16 bf16_t;
typedef __attribute__((ext_vector_type(16))) bf16_t bf16x16;
typedef __attribute__((ext_vector_type(8)))  float  f32x8;
typedef __attribute__((ext_vector_type(4)))  unsigned int u32x4;

union Frag {
  bf16x16 v;
  u32x4   q[2];
};

// ---- CDNA5 async global->LDS via inline asm (VGLOBAL opcode 98, GV mode) ----
// VDST = LDS byte address (low 32 bits of generic LDS pointer), VADDR = 64-bit
// global address. Tracked with ASYNCcnt.
__device__ __forceinline__ void stage16(const void* g, void* l) {
  unsigned int       la = (unsigned int)(unsigned long long)l;
  unsigned long long ga = (unsigned long long)g;
  asm volatile("global_load_async_to_lds_b128 %0, %1, off"
               :
               : "v"(la), "v"(ga)
               : "memory");
}

#if defined(__has_builtin)
#  if __has_builtin(__builtin_amdgcn_s_wait_asynccnt)
#    define HAS_WAIT_ASYNC 1
#  endif
#endif

__device__ __forceinline__ void wait_async_16() {
#ifdef HAS_WAIT_ASYNC
  __builtin_amdgcn_s_wait_asynccnt(16);
#else
  asm volatile("s_wait_asynccnt 16" ::: "memory");
#endif
}
__device__ __forceinline__ void wait_async_0() {
#ifdef HAS_WAIT_ASYNC
  __builtin_amdgcn_s_wait_asynccnt(0);
#else
  asm volatile("s_wait_asynccnt 0" ::: "memory");
#endif
}

__device__ __forceinline__ unsigned short f2bf(float f) {
  unsigned int u = __builtin_bit_cast(unsigned int, f);
  u += 0x7FFFu + ((u >> 16) & 1u);           // round-to-nearest-even
  return (unsigned short)(u >> 16);
}
__device__ __forceinline__ float bf2f(unsigned short h) {
  unsigned int u = ((unsigned int)h) << 16;
  return __builtin_bit_cast(float, u);
}

__device__ __forceinline__ f32x8 wmma_bf16(const Frag& a, const Frag& b, f32x8 c) {
  return __builtin_amdgcn_wmma_f32_16x16x32_bf16(
      /*neg_a=*/false, a.v, /*neg_b=*/false, b.v,
      /*c_mod=*/(short)0, c, /*reuse_a=*/false, /*reuse_b=*/false);
}

// ---------------------------------------------------------------- conversions
__global__ void cvt_bf16_kernel(const float* __restrict__ src,
                                unsigned short* __restrict__ dst, long long n) {
  long long i = (long long)blockIdx.x * blockDim.x + threadIdx.x;
  long long stride = (long long)gridDim.x * blockDim.x;
  for (; i < n; i += stride) dst[i] = f2bf(src[i]);
}

// Embedding gather + bf16 convert: xe[r,k] = bf16(emb[x[r], k])
__global__ void embed_gather_kernel(const int* __restrict__ x,
                                    const float* __restrict__ emb,
                                    unsigned short* __restrict__ xe) {
  long long i = (long long)blockIdx.x * blockDim.x + threadIdx.x;
  if (i >= (long long)ROWS * EMB) return;
  int r = (int)(i >> 9);           // EMB == 512
  int k = (int)(i & (EMB - 1));
  xe[i] = f2bf(emb[(size_t)x[r] * EMB + k]);
}

// ------------------------------------------------ xin = xe @ W_ih^T + b_ih + b_hh
__global__ __launch_bounds__(256) void xin_gemm_kernel(
    const unsigned short* __restrict__ xe,    // [ROWS, EMB] bf16
    const unsigned short* __restrict__ Wih,   // [HID, EMB]  bf16
    const float* __restrict__ b_ih, const float* __restrict__ b_hh,
    float* __restrict__ xin) {                // [ROWS, HID] f32
  int w = threadIdx.x >> 5, lane = threadIdx.x & 31;
  int ml = lane & 15, h = lane >> 4;
  int mtile = blockIdx.x >> 3;
  int nb    = blockIdx.x & 7;
  int rbase = mtile * 16;
  int ncol  = nb * 128 + w * 16 + ml;

  const u32x4* Arow = (const u32x4*)(xe  + (size_t)(rbase + ml) * EMB);
  const u32x4* Brow = (const u32x4*)(Wih + (size_t)ncol * EMB);
  f32x8 c = {};
#pragma unroll
  for (int kt = 0; kt < EMB / 32; ++kt) {
    Frag a, b;
    a.q[0] = Arow[(kt * 32 + h * 8) >> 3];
    a.q[1] = Arow[(kt * 32 + 16 + h * 8) >> 3];
    int bb = (kt * 32 + h * 8) >> 3;
    b.q[0] = Brow[bb];
    b.q[1] = Brow[bb + 2];
    c = wmma_bf16(a, b, c);
  }
  float bias = b_ih[ncol] + b_hh[ncol];
#pragma unroll
  for (int j = 0; j < 8; ++j) {
    int row = rbase + j + 8 * h;
    xin[(size_t)row * HID + ncol] = c[j] + bias;
  }
}

// -------------------------------------- one recurrence step: h = tanh(xin_t + h @ W_hh^T)
__global__ __launch_bounds__(256) void rnn_step_kernel(
    unsigned short* __restrict__ hs,          // [BSZ, SEQ, HID] bf16
    const float* __restrict__ xin,            // [ROWS, HID] f32
    const unsigned short* __restrict__ Whh,   // [HID, HID] bf16
    int t) {
  int w = threadIdx.x >> 5, lane = threadIdx.x & 31;
  int ml = lane & 15, h = lane >> 4;
  int ncol = blockIdx.x * 128 + w * 16 + ml;

  f32x8 c0 = {}, c1 = {};
  if (t > 0) {
    const u32x4* A0 = (const u32x4*)(hs + ((size_t)ml * SEQ + (t - 1)) * HID);
    const u32x4* A1 = (const u32x4*)(hs + ((size_t)(16 + ml) * SEQ + (t - 1)) * HID);
    const u32x4* Brow = (const u32x4*)(Whh + (size_t)ncol * HID);
#pragma unroll
    for (int kt = 0; kt < HID / 32; ++kt) {
      Frag a0, a1, b;
      a0.q[0] = A0[(kt * 32 + h * 8) >> 3];
      a0.q[1] = A0[(kt * 32 + 16 + h * 8) >> 3];
      a1.q[0] = A1[(kt * 32 + h * 8) >> 3];
      a1.q[1] = A1[(kt * 32 + 16 + h * 8) >> 3];
      int bb = (kt * 32 + h * 8) >> 3;
      b.q[0] = Brow[bb];
      b.q[1] = Brow[bb + 2];
      c0 = wmma_bf16(a0, b, c0);
      c1 = wmma_bf16(a1, b, c1);
    }
  }
#pragma unroll
  for (int j = 0; j < 8; ++j) {
    int b0 = j + 8 * h, b1 = 16 + j + 8 * h;
    size_t i0 = ((size_t)b0 * SEQ + t) * HID + ncol;
    size_t i1 = ((size_t)b1 * SEQ + t) * HID + ncol;
    hs[i0] = f2bf(tanhf(xin[i0] + c0[j]));
    hs[i1] = f2bf(tanhf(xin[i1] + c1[j]));
  }
}

// -------------------------------------- label logit: out[r] = hs[r,:]·W_cls[label[r],:] + b_cls
__global__ __launch_bounds__(256) void label_logit_kernel(
    const unsigned short* __restrict__ hs, const unsigned short* __restrict__ Wc,
    const float* __restrict__ b_cls, const int* __restrict__ label,
    float* __restrict__ out) {
  int w = threadIdx.x >> 5, lane = threadIdx.x & 31;
  int r = blockIdx.x * 8 + w;
  int lab = label[r];
  const unsigned short* hr = hs + (size_t)r * HID;
  const unsigned short* wr = Wc + (size_t)lab * HID;
  float acc = 0.f;
  for (int k = lane; k < HID; k += 32) acc += bf2f(hr[k]) * bf2f(wr[k]);
#pragma unroll
  for (int d = 16; d > 0; d >>= 1) acc += __shfl_xor(acc, d, 32);
  if (lane == 0) out[r] = acc + b_cls[lab];
}

// -------------------------------------- classifier GEMM fused with online logsumexp
// Block owns 64 rows; A (64x1024 bf16) register-resident (32 frags/wave).
// W_cls streamed in 32-col x 1024-K slabs, double-buffered in 2x64KB dynamic LDS
// via global_load_async_to_lds_b128 (ASYNCcnt-synchronized).
__global__ __launch_bounds__(256) void classifier_lse_kernel(
    const unsigned short* __restrict__ hs,    // [ROWS, HID] bf16
    const unsigned short* __restrict__ Wc,    // [VOCAB, HID] bf16
    const float* __restrict__ b_cls,          // [VOCAB]
    float* __restrict__ lse) {                // [ROWS]
  extern __shared__ __align__(16) char smem[];          // 2 x 64 KB
  unsigned short* Bs0 = (unsigned short*)smem;
  unsigned short* Bs1 = (unsigned short*)(smem + 32 * HID * 2);

  int tid = threadIdx.x, lane = tid & 31, w = tid >> 5;
  int mtile = w & 3, nsub = w >> 2;
  int ml = lane & 15, h = lane >> 4;
  int rbase = blockIdx.x * 64;

  Frag a[HID / 32];
  const u32x4* Arow = (const u32x4*)(hs + (size_t)(rbase + mtile * 16 + ml) * HID);
#pragma unroll
  for (int kt = 0; kt < HID / 32; ++kt) {
    a[kt].q[0] = Arow[(kt * 32 + h * 8) >> 3];
    a[kt].q[1] = Arow[(kt * 32 + 16 + h * 8) >> 3];
  }

  float mrun[8], srun[8];
#pragma unroll
  for (int j = 0; j < 8; ++j) { mrun[j] = -3.0e38f; srun[j] = 0.f; }

  int ncol_local = nsub * 16 + ml;

  // Prologue: stage slab 0 into Bs0.
  {
    const char* g = (const char*)Wc;
    char* l = (char*)Bs0;
#pragma unroll
    for (int i = 0; i < 16; ++i) {
      int off = (tid + 256 * i) * 16;
      stage16(g + off, l + off);
    }
  }

  for (int nb = 0; nb < VOCAB; nb += 32) {
    unsigned short* cur = ((nb >> 5) & 1) ? Bs1 : Bs0;
    unsigned short* nxt = ((nb >> 5) & 1) ? Bs0 : Bs1;

    if (nb > 0) __syncthreads();            // all waves done reading `nxt`
    if (nb + 32 < VOCAB) {                  // prefetch next slab (async)
      const char* g = (const char*)(Wc + (size_t)(nb + 32) * HID);
      char* l = (char*)nxt;
#pragma unroll
      for (int i = 0; i < 16; ++i) {
        int off = (tid + 256 * i) * 16;
        stage16(g + off, l + off);
      }
      wait_async_16();                      // my current-slab loads complete
    } else {
      wait_async_0();
    }
    __syncthreads();                        // everyone's current-slab loads complete

    f32x8 c = {};
    const u32x4* Bcol = (const u32x4*)(cur + (size_t)ncol_local * HID);
#pragma unroll
    for (int kt = 0; kt < HID / 32; ++kt) {
      Frag b;
      int bb = (kt * 32 + h * 8) >> 3;
      b.q[0] = Bcol[bb];
      b.q[1] = Bcol[bb + 2];
      c = wmma_bf16(a[kt], b, c);
    }
    float bias = b_cls[nb + ncol_local];
#pragma unroll
    for (int j = 0; j < 8; ++j) {
      float v = c[j] + bias;
      float mt = v;
#pragma unroll
      for (int d = 1; d < 16; d <<= 1) mt = fmaxf(mt, __shfl_xor(mt, d, 32));
      float p = __expf(v - mt);
#pragma unroll
      for (int d = 1; d < 16; d <<= 1) p += __shfl_xor(p, d, 32);
      float mo = mrun[j];
      float mn = fmaxf(mo, mt);
      srun[j] = srun[j] * __expf(mo - mn) + p * __expf(mt - mn);
      mrun[j] = mn;
    }
  }

  __syncthreads();
  float* msh = (float*)Bs0;        // overlay [2][64] stats on dead slab
  float* ssh = msh + 128;
  if (ml == 0) {
#pragma unroll
    for (int j = 0; j < 8; ++j) {
      int rl = mtile * 16 + j + 8 * h;
      msh[nsub * 64 + rl] = mrun[j];
      ssh[nsub * 64 + rl] = srun[j];
    }
  }
  __syncthreads();
  if (tid < 64) {
    float m0 = msh[tid],      s0 = ssh[tid];
    float m1 = msh[64 + tid], s1 = ssh[64 + tid];
    float mn = fmaxf(m0, m1);
    float s = s0 * __expf(m0 - mn) + s1 * __expf(m1 - mn);
    lse[rbase + tid] = mn + logf(s);
  }
}

// -------------------------------------- final mean NLL
__global__ void mean_nll_kernel(const float* __restrict__ lse,
                                const float* __restrict__ ll,
                                float* __restrict__ out) {
  __shared__ float red[256];
  float s = 0.f;
  for (int i = threadIdx.x; i < ROWS; i += 256) s += lse[i] - ll[i];
  red[threadIdx.x] = s;
  __syncthreads();
  for (int d = 128; d > 0; d >>= 1) {
    if ((int)threadIdx.x < d) red[threadIdx.x] += red[threadIdx.x + d];
    __syncthreads();
  }
  if (threadIdx.x == 0) out[0] = red[0] / (float)ROWS;
}

// ---------------------------------------------------------------- launcher
extern "C" void kernel_launch(void* const* d_in, const int* in_sizes, int n_in,
                              void* d_out, int out_size, void* d_ws, size_t ws_size,
                              hipStream_t stream) {
  (void)in_sizes; (void)n_in; (void)out_size; (void)ws_size;

  const int*   x     = (const int*)d_in[0];
  const int*   label = (const int*)d_in[1];
  const float* emb   = (const float*)d_in[2];
  const float* W_ih  = (const float*)d_in[3];
  const float* W_hh  = (const float*)d_in[4];
  const float* b_ih  = (const float*)d_in[5];
  const float* b_hh  = (const float*)d_in[6];
  const float* W_cls = (const float*)d_in[7];
  const float* b_cls = (const float*)d_in[8];

  char* ws = (char*)d_ws;
  auto alloc = [&](size_t bytes) {
    char* p = ws;
    ws += (bytes + 255) & ~(size_t)255;
    return p;
  };
  unsigned short* Wih_bf  = (unsigned short*)alloc((size_t)HID * EMB * 2);
  unsigned short* Whh_bf  = (unsigned short*)alloc((size_t)HID * HID * 2);
  unsigned short* Wcls_bf = (unsigned short*)alloc((size_t)VOCAB * HID * 2);
  unsigned short* xe_bf   = (unsigned short*)alloc((size_t)ROWS * EMB * 2);
  float*          xin     = (float*)alloc((size_t)ROWS * HID * 4);
  unsigned short* hs      = (unsigned short*)alloc((size_t)ROWS * HID * 2);
  float*          lse     = (float*)alloc((size_t)ROWS * 4);
  float*          labl    = (float*)alloc((size_t)ROWS * 4);

  cvt_bf16_kernel<<<1024, 256, 0, stream>>>(W_ih, Wih_bf, (long long)HID * EMB);
  cvt_bf16_kernel<<<2048, 256, 0, stream>>>(W_hh, Whh_bf, (long long)HID * HID);
  cvt_bf16_kernel<<<8192, 256, 0, stream>>>(W_cls, Wcls_bf, (long long)VOCAB * HID);
  embed_gather_kernel<<<(ROWS * EMB) / 256, 256, 0, stream>>>(x, emb, xe_bf);

  xin_gemm_kernel<<<512 * 8, 256, 0, stream>>>(xe_bf, Wih_bf, b_ih, b_hh, xin);

  for (int t = 0; t < SEQ; ++t)
    rnn_step_kernel<<<8, 256, 0, stream>>>(hs, xin, Whh_bf, t);

  label_logit_kernel<<<ROWS / 8, 256, 0, stream>>>(hs, Wcls_bf, b_cls, label, labl);
  classifier_lse_kernel<<<ROWS / 64, 256, 2 * 32 * HID * 2, stream>>>(
      hs, Wcls_bf, b_cls, lse);
  mean_nll_kernel<<<1, 256, 0, stream>>>(lse, labl, (float*)d_out);
}
